// CausalSelfAttention_36807869727034
// MI455X (gfx1250) — matrix operand order repaired and checked by reading, not compiled, not run on hardware
//
#include <hip/hip_runtime.h>
#include <hip/hip_bf16.h>

#define TT 2048
#define CC 1024
#define HH 16
#define HD 64

typedef __bf16 bf16;
typedef __attribute__((ext_vector_type(16))) __bf16 v16bf;
typedef __attribute__((ext_vector_type(8)))  float  v8f;
typedef __attribute__((ext_vector_type(4)))  unsigned int v4u;
typedef __attribute__((ext_vector_type(4)))  float  v4f;

union V16BF { v16bf v; v4u u[2]; };

// ---- WMMA fragment loaders (16x16x32 bf16, contiguous k order used for both A and B) -------------------

// A (16x32, MxK), row-major source with leading dim ld.
// lanes 0-15: row=lane,    elems = K0..15
// lanes16-31: row=lane-16, elems = K16..31
__device__ __forceinline__ v16bf load_a_frag(const bf16* base, int ld, int m0, int k0) {
  int lane = threadIdx.x & 31;
  const bf16* p = base + (long)(m0 + (lane & 15)) * ld + k0 + ((lane >> 4) << 4);
  V16BF r;
  r.u[0] = *(const v4u*)p;
  r.u[1] = *(const v4u*)(p + 8);
  return r.v;
}

// B (32x16, KxN) where B[k][n] = W[n][k], W row-major with leading dim ld.
// lanes 0-15: col=lane,    elems = K k0..k0+15 (contiguous in W row)
// lanes16-31: col=lane-16, elems = K k0+16..k0+31
__device__ __forceinline__ v16bf load_bT_frag(const bf16* base, int ld, int n0, int k0) {
  int lane = threadIdx.x & 31;
  const bf16* p = base + (long)(n0 + (lane & 15)) * ld + k0 + ((lane >> 4) << 4);
  V16BF r;
  r.u[0] = *(const v4u*)p;
  r.u[1] = *(const v4u*)(p + 8);
  return r.v;
}

__device__ __forceinline__ v8f wmma_bf16(v16bf a, v16bf b, v8f c) {
  return __builtin_amdgcn_wmma_f32_16x16x32_bf16(false, a, false, b, (short)0, c,
                                                 false, false);
}

// ---- fp32 -> bf16 conversion ----------------------------------------------

__global__ __launch_bounds__(256) void cvt_f32_bf16(const float* __restrict__ in,
                                                    bf16* __restrict__ out, int n) {
  int i = (blockIdx.x * 256 + threadIdx.x) * 4;
  if (i + 3 < n) {
    v4f v = *(const v4f*)(in + i);
    out[i + 0] = (bf16)v.x;
    out[i + 1] = (bf16)v.y;
    out[i + 2] = (bf16)v.z;
    out[i + 3] = (bf16)v.w;
  }
}

// ---- QKV GEMM: (4096x1024) @ W_attn^T + b -> Q,K (B,H,T,64) and V^T (B,H,64,T)

__global__ __launch_bounds__(256) void qkv_gemm(const bf16* __restrict__ X,
                                                const bf16* __restrict__ W,
                                                const float* __restrict__ bias,
                                                bf16* __restrict__ Q,
                                                bf16* __restrict__ Kc,
                                                bf16* __restrict__ Vt) {
  int w = threadIdx.x >> 5;
  int lane = threadIdx.x & 31;
  int m0 = blockIdx.y * 128 + (w >> 1) * 32;
  int n0 = blockIdx.x * 64 + (w & 1) * 32;
  v8f acc[2][2] = {};
  for (int k0 = 0; k0 < CC; k0 += 32) {
    v16bf a0 = load_a_frag(X, CC, m0, k0);
    v16bf a1 = load_a_frag(X, CC, m0 + 16, k0);
    v16bf b0 = load_bT_frag(W, CC, n0, k0);
    v16bf b1 = load_bT_frag(W, CC, n0 + 16, k0);
    acc[0][0] = wmma_bf16(a0, b0, acc[0][0]);
    acc[0][1] = wmma_bf16(a0, b1, acc[0][1]);
    acc[1][0] = wmma_bf16(a1, b0, acc[1][0]);
    acc[1][1] = wmma_bf16(a1, b1, acc[1][1]);
  }
  int hi = (lane >> 4) << 3;
  for (int i = 0; i < 2; ++i)
    for (int j = 0; j < 2; ++j) {
      int n = n0 + j * 16 + (lane & 15);
      int which = n >> 10;
      int rem = n & 1023;
      int h = rem >> 6, d = rem & 63;
      float bv = bias[n];
      for (int r = 0; r < 8; ++r) {
        int m = m0 + i * 16 + r + hi;
        int b = m >> 11, t = m & 2047;
        bf16 val = (bf16)(acc[i][j][r] + bv);
        long bh = (long)(b * HH + h);
        if (which == 0)      Q [bh * TT * HD + (long)t * HD + d] = val;
        else if (which == 1) Kc[bh * TT * HD + (long)t * HD + d] = val;
        else                 Vt[bh * HD * TT + (long)d * TT + t] = val;
      }
    }
}

// ---- Flash attention: 1 wave = 16 query rows, 32-key tiles, online softmax --

__global__ __launch_bounds__(128) void attn(const bf16* __restrict__ Q,
                                            const bf16* __restrict__ Kc,
                                            const bf16* __restrict__ Vt,
                                            bf16* __restrict__ Y) {
  __shared__ __align__(16) bf16 ldsP[4][16 * 32];
  int w = threadIdx.x >> 5;
  int lane = threadIdx.x & 31;
  int bh = blockIdx.y;                 // 0..31
  int b = bh >> 4, h = bh & 15;
  int q0 = blockIdx.x * 64 + w * 16;
  const bf16* qb = Q  + (long)bh * TT * HD;
  const bf16* kb = Kc + (long)bh * TT * HD;
  const bf16* vb = Vt + (long)bh * HD * TT;

  v16bf qa0 = load_a_frag(qb, HD, q0, 0);
  v16bf qa1 = load_a_frag(qb, HD, q0, 32);
  v8f o[4] = {};
  float mrow[8], lrow[8];
  for (int r = 0; r < 8; ++r) { mrow[r] = -3.0e38f; lrow[r] = 0.0f; }
  int hi = (lane >> 4) << 3;

  int nkt = ((q0 + 15) >> 5) + 1;      // causal: keys up to q0+15
  for (int kt = 0; kt < nkt; ++kt) {
    int key0 = kt * 32;
    v8f s0 = {}, s1 = {};
    s0 = wmma_bf16(qa0, load_bT_frag(kb, HD, key0, 0),       s0);
    s0 = wmma_bf16(qa1, load_bT_frag(kb, HD, key0, 32),      s0);
    s1 = wmma_bf16(qa0, load_bT_frag(kb, HD, key0 + 16, 0),  s1);
    s1 = wmma_bf16(qa1, load_bT_frag(kb, HD, key0 + 16, 32), s1);
    int c0 = key0 + (lane & 15);
    int c1 = c0 + 16;
    for (int r = 0; r < 8; ++r) {
      int qr = q0 + r + hi;
      float v0 = (c0 <= qr) ? s0[r] * 0.125f : -3.0e38f;
      float v1 = (c1 <= qr) ? s1[r] * 0.125f : -3.0e38f;
      float mx = fmaxf(v0, v1);
      for (int off = 8; off; off >>= 1) mx = fmaxf(mx, __shfl_xor(mx, off, 16));
      float mnew = fmaxf(mrow[r], mx);
      float alpha = __expf(mrow[r] - mnew);
      float p0 = __expf(v0 - mnew);
      float p1 = __expf(v1 - mnew);
      float rs = p0 + p1;
      for (int off = 8; off; off >>= 1) rs += __shfl_xor(rs, off, 16);
      lrow[r] = lrow[r] * alpha + rs;
      mrow[r] = mnew;
      for (int dt = 0; dt < 4; ++dt) o[dt][r] *= alpha;
      // acc layout -> A-fragment layout via LDS (wave-private 1KB tile)
      ldsP[w][(r + hi) * 32 + (lane & 15)]      = (bf16)p0;
      ldsP[w][(r + hi) * 32 + (lane & 15) + 16] = (bf16)p1;
    }
    v16bf pf = load_a_frag(&ldsP[w][0], 32, 0, 0);
    for (int dt = 0; dt < 4; ++dt)
      o[dt] = wmma_bf16(pf, load_bT_frag(vb, TT, dt * 16, key0), o[dt]);
  }

  for (int dt = 0; dt < 4; ++dt) {
    int d = h * 64 + dt * 16 + (lane & 15);
    for (int r = 0; r < 8; ++r) {
      int qr = q0 + r + hi;
      Y[(long)(b * TT + qr) * CC + d] = (bf16)(o[dt][r] / lrow[r]);
    }
  }
}

// ---- Output projection: Y(4096x1024) @ W_proj^T + b -> fp32 ----------------

__global__ __launch_bounds__(256) void proj_gemm(const bf16* __restrict__ Yb,
                                                 const bf16* __restrict__ W,
                                                 const float* __restrict__ bias,
                                                 float* __restrict__ out) {
  int w = threadIdx.x >> 5;
  int lane = threadIdx.x & 31;
  int m0 = blockIdx.y * 128 + (w >> 1) * 32;
  int n0 = blockIdx.x * 64 + (w & 1) * 32;
  v8f acc[2][2] = {};
  for (int k0 = 0; k0 < CC; k0 += 32) {
    v16bf a0 = load_a_frag(Yb, CC, m0, k0);
    v16bf a1 = load_a_frag(Yb, CC, m0 + 16, k0);
    v16bf b0 = load_bT_frag(W, CC, n0, k0);
    v16bf b1 = load_bT_frag(W, CC, n0 + 16, k0);
    acc[0][0] = wmma_bf16(a0, b0, acc[0][0]);
    acc[0][1] = wmma_bf16(a0, b1, acc[0][1]);
    acc[1][0] = wmma_bf16(a1, b0, acc[1][0]);
    acc[1][1] = wmma_bf16(a1, b1, acc[1][1]);
  }
  int hi = (lane >> 4) << 3;
  for (int i = 0; i < 2; ++i)
    for (int j = 0; j < 2; ++j) {
      int n = n0 + j * 16 + (lane & 15);
      float bv = bias[n];
      for (int r = 0; r < 8; ++r)
        out[(long)(m0 + i * 16 + r + hi) * CC + n] = acc[i][j][r] + bv;
    }
}

// ---- host-side launch -------------------------------------------------------

extern "C" void kernel_launch(void* const* d_in, const int* in_sizes, int n_in,
                              void* d_out, int out_size, void* d_ws, size_t ws_size,
                              hipStream_t stream) {
  (void)in_sizes; (void)n_in; (void)out_size; (void)ws_size;
  const float* x      = (const float*)d_in[0];
  const float* W_attn = (const float*)d_in[1];
  const float* b_attn = (const float*)d_in[2];
  const float* W_proj = (const float*)d_in[3];
  const float* b_proj = (const float*)d_in[4];

  char* ws = (char*)d_ws;
  const size_t NX  = 4194304;  // B*T*C
  const size_t NWA = 3145728;  // 3C*C
  const size_t NWP = 1048576;  // C*C
  bf16* xb  = (bf16*)(ws);                       // 8 MiB
  bf16* Wab = (bf16*)(ws + 8388608);             // 6 MiB
  bf16* Wpb = (bf16*)(ws + 8388608 + 6291456);   // 2 MiB
  bf16* Qb  = (bf16*)(ws + 16777216);            // 8 MiB
  bf16* Kb  = (bf16*)(ws + 16777216 + 8388608);  // 8 MiB
  bf16* Vtb = (bf16*)(ws + 16777216 + 2 * 8388608);
  bf16* Yb  = (bf16*)(ws + 16777216 + 3 * 8388608);

  cvt_f32_bf16<<<NX  / 1024, 256, 0, stream>>>(x,      xb,  (int)NX);
  cvt_f32_bf16<<<NWA / 1024, 256, 0, stream>>>(W_attn, Wab, (int)NWA);
  cvt_f32_bf16<<<NWP / 1024, 256, 0, stream>>>(W_proj, Wpb, (int)NWP);

  qkv_gemm<<<dim3(48, 32), 256, 0, stream>>>(xb, Wab, b_attn, Qb, Kb, Vtb);
  attn    <<<dim3(32, 32), 128, 0, stream>>>(Qb, Kb, Vtb, Yb);
  proj_gemm<<<dim3(16, 32), 256, 0, stream>>>(Yb, Wpb, b_proj, (float*)d_out);
}
